// MultiSlotOuterModel_89043261981281
// MI455X (gfx1250) — matrix operand order repaired and checked by reading, not compiled, not run on hardware
//
#include <hip/hip_runtime.h>

// ---------------------------------------------------------------------------
// Problem constants (match reference)
// ---------------------------------------------------------------------------
#define BATCH    256
#define DIM      1024
#define NSLOT    200000
#define KDIM     64
#define NGROUPS  (NSLOT / 64)   // 3125 groups of 64 slots (200000 % 64 == 0)
#define NCHUNK   256            // N-split factor (grid of flash kernel)
#define LOG2E    1.44269504088896340736f

typedef __attribute__((ext_vector_type(16))) __bf16 v16bf;
typedef __attribute__((ext_vector_type(8)))  float  v8f;

// float -> bf16, round-to-nearest-even
__device__ __forceinline__ unsigned short f2bf(float f) {
    unsigned int u = __float_as_uint(f);
    unsigned int r = u + 0x7fffu + ((u >> 16) & 1u);
    return (unsigned short)(r >> 16);
}
// float -> bf16 truncate (cheap, used for P tile only)
__device__ __forceinline__ unsigned short f2bf_trunc(float f) {
    return (unsigned short)(__float_as_uint(f) >> 16);
}

// 16 contiguous bf16 halves -> fragment (two b128 loads)
__device__ __forceinline__ v16bf ld_frag16(const unsigned short* p) {
    union { uint4 u[2]; v16bf v; } f;
    f.u[0] = *(const uint4*)(p);
    f.u[1] = *(const uint4*)(p + 8);
    return f.v;
}
// two separated 8-half runs -> fragment
__device__ __forceinline__ v16bf ld_frag_split(const unsigned short* p0,
                                               const unsigned short* p1) {
    union { uint4 u[2]; v16bf v; } f;
    f.u[0] = *(const uint4*)(p0);
    f.u[1] = *(const uint4*)(p1);
    return f.v;
}

__device__ __forceinline__ v8f wmma_bf16(v16bf a, v16bf b, v8f c) {
    return __builtin_amdgcn_wmma_f32_16x16x32_bf16(
        false, a, false, b, (short)0, c, false, false);
}

__device__ __forceinline__ float fast_exp2(float x) {
#if __has_builtin(__builtin_amdgcn_exp2f)
    return __builtin_amdgcn_exp2f(x);
#else
    return exp2f(x);
#endif
}

// max over the 16-lane group (per wave32 half), pure VALU (no LDS) when
// dpp8/permlane16 builtins are available.
#define DPP8_SEL(a,b,c,d,e,f,g,h) \
    ((a)|((b)<<3)|((c)<<6)|((d)<<9)|((e)<<12)|((f)<<15)|((g)<<18)|((h)<<21))

__device__ __forceinline__ float rowmax16(float x) {
#if __has_builtin(__builtin_amdgcn_mov_dpp8) && __has_builtin(__builtin_amdgcn_permlane16)
    x = fmaxf(x, __int_as_float(__builtin_amdgcn_mov_dpp8(
            __float_as_int(x), DPP8_SEL(1,0,3,2,5,4,7,6))));   // xor 1
    x = fmaxf(x, __int_as_float(__builtin_amdgcn_mov_dpp8(
            __float_as_int(x), DPP8_SEL(2,3,0,1,6,7,4,5))));   // xor 2
    x = fmaxf(x, __int_as_float(__builtin_amdgcn_mov_dpp8(
            __float_as_int(x), DPP8_SEL(4,5,6,7,0,1,2,3))));   // xor 4
    x = fmaxf(x, __int_as_float(__builtin_amdgcn_permlane16(   // xor 8
            __float_as_int(x), __float_as_int(x),
            0xFEDCBA98u, 0x76543210u, false, false)));
#else
    x = fmaxf(x, __shfl_xor(x, 1, 32));
    x = fmaxf(x, __shfl_xor(x, 2, 32));
    x = fmaxf(x, __shfl_xor(x, 4, 32));
    x = fmaxf(x, __shfl_xor(x, 8, 32));
#endif
    return x;
}

// ---------------------------------------------------------------------------
// Kernel 1: cue_outer[b,k] = log2(e) * sum_d cue[b,d]*w_cue[k,d]  (bf16)
//           (log2e folded in so the flash softmax runs in base-2 domain)
// ---------------------------------------------------------------------------
__global__ __launch_bounds__(256) void cueproj_kernel(
    const float* __restrict__ cue, const float* __restrict__ wcue,
    unsigned short* __restrict__ cueb)
{
    int idx = blockIdx.x * 256 + threadIdx.x;      // 0 .. 16383
    int b = idx >> 6, k = idx & 63;
    const float4* cr = (const float4*)(cue  + (size_t)b * DIM);
    const float4* wr = (const float4*)(wcue + (size_t)k * DIM);
    float acc = 0.f;
    #pragma unroll 4
    for (int i = 0; i < DIM / 4; ++i) {
        float4 a = cr[i], w = wr[i];
        acc += a.x * w.x + a.y * w.y + a.z * w.z + a.w * w.w;
    }
    cueb[idx] = f2bf(acc * LOG2E);
}

// ---------------------------------------------------------------------------
// Kernel 2: fused flash pass. grid = NCHUNK x 512 threads (16 waves).
// Wave w owns batch rows [16w,16w+16) -> the 16 waves cover all 256 rows and
// share one 64-slot tile per iteration (slot_matrix is read exactly once
// device-wide). Priority is folded into the V tile and into a "prio column"
// WMMA that accumulates l with the same online rescale as the output.
// ---------------------------------------------------------------------------
__device__ __forceinline__ void stage_tile(
    const float* __restrict__ slot, const float* __restrict__ prio, long g,
    unsigned short (*sA)[72], unsigned short (*sB)[72],
    unsigned short* sPr, int tid)
{
    int s  = tid >> 3;             // local slot 0..63
    int f0 = (tid & 7) << 3;       // feature 0..56 step 8
    const float* base = slot + ((size_t)g * 64 + s) * KDIM + f0;
    const float4 v0 = *(const float4*)(base);
    const float4 v1 = *(const float4*)(base + 4);
    const float  pr = prio[(size_t)g * 64 + s];

    // sA: [slot][feat], raw values (sim B operand)
    uint4 pk;
    pk.x = (unsigned int)f2bf(v0.x) | ((unsigned int)f2bf(v0.y) << 16);
    pk.y = (unsigned int)f2bf(v0.z) | ((unsigned int)f2bf(v0.w) << 16);
    pk.z = (unsigned int)f2bf(v1.x) | ((unsigned int)f2bf(v1.y) << 16);
    pk.w = (unsigned int)f2bf(v1.z) | ((unsigned int)f2bf(v1.w) << 16);
    *(uint4*)&sA[s][f0] = pk;

    // sB: [feat][slot], priority-scaled values (PV B operand)
    sB[f0 + 0][s] = f2bf(v0.x * pr);
    sB[f0 + 1][s] = f2bf(v0.y * pr);
    sB[f0 + 2][s] = f2bf(v0.z * pr);
    sB[f0 + 3][s] = f2bf(v0.w * pr);
    sB[f0 + 4][s] = f2bf(v1.x * pr);
    sB[f0 + 5][s] = f2bf(v1.y * pr);
    sB[f0 + 6][s] = f2bf(v1.z * pr);
    sB[f0 + 7][s] = f2bf(v1.w * pr);

    // sPr: priority vector (bf16) for the l-accumulating WMMA
    if (tid < 64) sPr[tid] = f2bf(prio[(size_t)g * 64 + tid]);
}

__global__ __launch_bounds__(512) void flash_kernel(
    const float* __restrict__ slot, const float* __restrict__ prio,
    const unsigned short* __restrict__ cueb,
    float* __restrict__ mws, float* __restrict__ lws, float* __restrict__ accws)
{
    // Rows padded to 72 halves (144 B): 16B-aligned rows, bank-spread strides.
    __shared__ __align__(16) unsigned short sA[2][64][72]; // [slot][feat]
    __shared__ __align__(16) unsigned short sB[2][64][72]; // [feat][slot*prio]
    __shared__ __align__(16) unsigned short sP[16][16][72];// per-wave P tile
    __shared__ __align__(16) unsigned short sPr[2][72];    // prio bf16

    const int tid  = threadIdx.x;
    const int wv   = tid >> 5;
    const int ln   = tid & 31;
    const int half = ln >> 4;
    const int lm   = ln & 15;
    const int chunk = blockIdx.x;

    // A fragments of cue_outer (rows wv*16 + lm), loaded once.
    const unsigned short* crow = cueb + (size_t)(wv * 16 + lm) * KDIM;
    const v16bf a0 = ld_frag_split(crow +      8 * half, crow + 16 + 8 * half);
    const v16bf a1 = ld_frag_split(crow + 32 + 8 * half, crow + 48 + 8 * half);

    float m8[8];
    v8f acc[5];                     // acc[0..3] = retrieved tiles, acc[4] = l
    #pragma unroll
    for (int r = 0; r < 8; ++r) m8[r] = -1e30f;
    #pragma unroll
    for (int t = 0; t < 5; ++t) {
        v8f z;
        #pragma unroll
        for (int r = 0; r < 8; ++r) z[r] = 0.f;
        acc[t] = z;
    }

    long g = chunk;
    stage_tile(slot, prio, g, sA[0], sB[0], sPr[0], tid);
    int buf = 0;

    for (; g < NGROUPS; g += NCHUNK) {
        __syncthreads();
        long gn = g + NCHUNK;
        if (gn < NGROUPS)
            stage_tile(slot, prio, gn, sA[buf ^ 1], sB[buf ^ 1], sPr[buf ^ 1], tid);

        // ---- sim: 16 rows x 64 slots (4 col tiles x 2 chained K-steps) -----
        v8f c[4];
        #pragma unroll
        for (int j = 0; j < 4; ++j) {
            v8f z;
            #pragma unroll
            for (int r = 0; r < 8; ++r) z[r] = 0.f;
            v16bf bk0 = ld_frag_split(&sA[buf][16 * j + lm][     8 * half], &sA[buf][16 * j + lm][16 + 8 * half]);
            v16bf bk1 = ld_frag_split(&sA[buf][16 * j + lm][32 + 8 * half], &sA[buf][16 * j + lm][48 + 8 * half]);
            z = wmma_bf16(a0, bk0, z);
            z = wmma_bf16(a1, bk1, z);
            c[j] = z;
        }

        // ---- online softmax: row max (VALU-only reduction), rescale --------
        float mn8[8];
        #pragma unroll
        for (int r = 0; r < 8; ++r) {
            float t = fmaxf(fmaxf(c[0][r], c[1][r]), fmaxf(c[2][r], c[3][r]));
            t = rowmax16(t);
            float mn = fmaxf(m8[r], t);
            float sc = fast_exp2(m8[r] - mn);
            m8[r] = mn;
            mn8[r] = mn;
            acc[0][r] *= sc; acc[1][r] *= sc; acc[2][r] *= sc;
            acc[3][r] *= sc; acc[4][r] *= sc;
        }

        // ---- P = exp2(sim - m), repack C-layout -> A-fragment via LDS ------
        #pragma unroll
        for (int j = 0; j < 4; ++j) {
            #pragma unroll
            for (int r = 0; r < 8; ++r) {
                sP[wv][r + 8 * half][16 * j + lm] =
                    f2bf_trunc(fast_exp2(c[j][r] - mn8[r]));
            }
        }
        __builtin_amdgcn_wave_barrier();   // intra-wave LDS ordering fence

        v16bf ap0 = ld_frag_split(&sP[wv][lm][     8 * half], &sP[wv][lm][16 + 8 * half]);
        v16bf ap1 = ld_frag_split(&sP[wv][lm][32 + 8 * half], &sP[wv][lm][48 + 8 * half]);

        // ---- acc += P(16x64) x (V*prio)(64x64): 8 WMMA ---------------------
        #pragma unroll
        for (int t = 0; t < 4; ++t) {
            v16bf bv0 = ld_frag_split(&sB[buf][lm + 16 * t][     8 * half], &sB[buf][lm + 16 * t][16 + 8 * half]);
            v16bf bv1 = ld_frag_split(&sB[buf][lm + 16 * t][32 + 8 * half], &sB[buf][lm + 16 * t][48 + 8 * half]);
            acc[t] = wmma_bf16(ap0, bv0, acc[t]);
            acc[t] = wmma_bf16(ap1, bv1, acc[t]);
        }
        // ---- l += P x prio-column (2 WMMA, broadcast B from sPr) -----------
        {
            v16bf bp0 = ld_frag_split(&sPr[buf][     8 * half], &sPr[buf][16 + 8 * half]);
            v16bf bp1 = ld_frag_split(&sPr[buf][32 + 8 * half], &sPr[buf][48 + 8 * half]);
            acc[4] = wmma_bf16(ap0, bp0, acc[4]);
            acc[4] = wmma_bf16(ap1, bp1, acc[4]);
        }
        __builtin_amdgcn_wave_barrier();

        buf ^= 1;
    }

    // ---- write partials: (m, l, acc[64]) per (chunk, row) -------------------
    const int rowbase = wv * 16 + 8 * half;
    if (lm == 0) {
        #pragma unroll
        for (int r = 0; r < 8; ++r) {
            mws[(size_t)chunk * BATCH + rowbase + r] = m8[r];
            lws[(size_t)chunk * BATCH + rowbase + r] = acc[4][r];
        }
    }
    #pragma unroll
    for (int t = 0; t < 4; ++t) {
        #pragma unroll
        for (int r = 0; r < 8; ++r) {
            accws[((size_t)chunk * BATCH + rowbase + r) * KDIM + t * 16 + lm] = acc[t][r];
        }
    }
}

// ---------------------------------------------------------------------------
// Kernel 3: merge chunk partials (base-2 log-sum-exp) -> retrieved (256 x 64)
// ---------------------------------------------------------------------------
__global__ __launch_bounds__(64) void reduce_kernel(
    const float* __restrict__ mws, const float* __restrict__ lws,
    const float* __restrict__ accws, float* __restrict__ retr)
{
    int b = blockIdx.x;        // batch row
    int t = threadIdx.x;       // feature
    float mg = -1e30f;
    for (int c = 0; c < NCHUNK; ++c)
        mg = fmaxf(mg, mws[(size_t)c * BATCH + b]);
    float lg = 0.f, ag = 0.f;
    for (int c = 0; c < NCHUNK; ++c) {
        float w = fast_exp2(mws[(size_t)c * BATCH + b] - mg);
        lg += w * lws[(size_t)c * BATCH + b];
        ag += w * accws[((size_t)c * BATCH + b) * KDIM + t];
    }
    retr[b * KDIM + t] = ag / fmaxf(lg, 1e-30f);
}

// ---------------------------------------------------------------------------
// Kernel 4: out[b,d] = sum_k retrieved[b,k] * w_dec[d,k]
// ---------------------------------------------------------------------------
__global__ __launch_bounds__(256) void decoder_kernel(
    const float* __restrict__ retr, const float* __restrict__ wdec,
    float* __restrict__ out)
{
    int idx = blockIdx.x * 256 + threadIdx.x;   // 0 .. 262143
    int b = idx >> 10, d = idx & 1023;
    const float4* rr = (const float4*)(retr + (size_t)b * KDIM);
    const float4* wr = (const float4*)(wdec + (size_t)d * KDIM);
    float acc = 0.f;
    #pragma unroll
    for (int i = 0; i < KDIM / 4; ++i) {
        float4 a = rr[i], w = wr[i];
        acc += a.x * w.x + a.y * w.y + a.z * w.z + a.w * w.w;
    }
    out[idx] = acc;
}

// ---------------------------------------------------------------------------
// Host launcher
// ---------------------------------------------------------------------------
extern "C" void kernel_launch(void* const* d_in, const int* in_sizes, int n_in,
                              void* d_out, int out_size, void* d_ws, size_t ws_size,
                              hipStream_t stream) {
    const float* cue  = (const float*)d_in[0];   // (256, 1024)
    const float* slot = (const float*)d_in[1];   // (200000, 64)
    const float* prio = (const float*)d_in[2];   // (200000,)
    const float* wcue = (const float*)d_in[3];   // (64, 1024)
    const float* wdec = (const float*)d_in[4];   // (1024, 64)
    float* out = (float*)d_out;                  // (256, 1024)

    char* w = (char*)d_ws;
    unsigned short* cueb = (unsigned short*)w;                       // 32 KB
    size_t off = (size_t)BATCH * KDIM * sizeof(unsigned short);
    float* mws   = (float*)(w + off); off += (size_t)NCHUNK * BATCH * sizeof(float);
    float* lws   = (float*)(w + off); off += (size_t)NCHUNK * BATCH * sizeof(float);
    float* accws = (float*)(w + off); off += (size_t)NCHUNK * BATCH * KDIM * sizeof(float);
    float* retr  = (float*)(w + off);                                // 64 KB

    cueproj_kernel<<<(BATCH * KDIM) / 256, 256, 0, stream>>>(cue, wcue, cueb);
    flash_kernel<<<NCHUNK, 512, 0, stream>>>(slot, prio, cueb, mws, lws, accws);
    reduce_kernel<<<BATCH, KDIM, 0, stream>>>(mws, lws, accws, retr);
    decoder_kernel<<<(BATCH * DIM) / 256, 256, 0, stream>>>(retr, wdec, out);
}
